// GuardedLayer_22943715295271
// MI455X (gfx1250) — matrix operand order repaired and checked by reading, not compiled, not run on hardware
//
#include <hip/hip_runtime.h>

typedef __attribute__((ext_vector_type(16))) __bf16 v16bf;
typedef __attribute__((ext_vector_type(8)))  __bf16 v8bf;
typedef __attribute__((ext_vector_type(4)))  __bf16 v4bf;
typedef __attribute__((ext_vector_type(8)))  float  v8f;

#define D_DIM  512
#define E_DIM  512
#define C_CASES 8
#define CP      16   // C*P score channels

#define BM 128
#define BN 128
#define BK 32
#define ASTR 48      // padded LDS row stride (bf16 elems): 96B, keeps 16B alignment, cuts conflicts

// -------------------- presence kernel --------------------
// presence[n,c] = prod_p sigmoid(x[n]·Wp[c,p] + bp[c,p]); zero if <= EPS
__global__ __launch_bounds__(256)
void presence_kernel(const float* __restrict__ x,
                     const float* __restrict__ Wp,
                     const float* __restrict__ bp,
                     float* __restrict__ presence)
{
    __shared__ float wp_s[CP * D_DIM];            // 32 KB
    const int tid = threadIdx.x;
    for (int i = tid * 4; i < CP * D_DIM; i += 256 * 4)
        *(float4*)&wp_s[i] = *(const float4*)&Wp[i];
    __syncthreads();

    const int lane = tid & 31;
    const int wave = tid >> 5;

    for (int r8 = 0; r8 < 8; ++r8) {
        const int row = blockIdx.x * 64 + wave * 8 + r8;
        float acc[CP];
#pragma unroll
        for (int j = 0; j < CP; ++j) acc[j] = 0.f;

#pragma unroll 4
        for (int i = 0; i < D_DIM / 32; ++i) {
            const int d = i * 32 + lane;
            const float xv = x[(size_t)row * D_DIM + d];
#pragma unroll
            for (int j = 0; j < CP; ++j)
                acc[j] = fmaf(xv, wp_s[j * D_DIM + d], acc[j]);
        }
#pragma unroll
        for (int j = 0; j < CP; ++j) {
#pragma unroll
            for (int off = 16; off > 0; off >>= 1)
                acc[j] += __shfl_xor(acc[j], off, 32);
        }
        if (lane < C_CASES) {
            const int c = lane;
            const float s0 = acc[c * 2 + 0] + bp[c * 2 + 0];
            const float s1 = acc[c * 2 + 1] + bp[c * 2 + 1];
            const float p0 = 1.f / (1.f + __expf(-s0));
            const float p1 = 1.f / (1.f + __expf(-s1));
            float pr = p0 * p1;
            if (pr <= 1e-4f) pr = 0.f;
            presence[(size_t)row * C_CASES + c] = pr;
        }
    }
}

// -------------------- Wc -> bf16 transpose (once) --------------------
// WcT[c][e][d] = (bf16) Wc[c][d][e]   (K-contiguous rows for direct B-fragment loads)
__global__ __launch_bounds__(256)
void wct_kernel(const float* __restrict__ Wc, __bf16* __restrict__ WcT)
{
    __shared__ float t[32][33];
    const int c  = blockIdx.z;
    const int e0 = blockIdx.x * 32;
    const int d0 = blockIdx.y * 32;
    const int tx = threadIdx.x & 31;
    const int ty = threadIdx.x >> 5;          // 0..7
#pragma unroll
    for (int i = 0; i < 4; ++i) {
        const int r = ty + i * 8;             // d-local
        t[r][tx] = Wc[(size_t)(c * D_DIM + d0 + r) * E_DIM + e0 + tx];
    }
    __syncthreads();
#pragma unroll
    for (int i = 0; i < 4; ++i) {
        const int r = ty + i * 8;             // e-local
        WcT[(size_t)(c * E_DIM + e0 + r) * D_DIM + d0 + tx] = (__bf16)t[tx][r];
    }
}

// -------------------- fused guarded GEMM --------------------
// out[n,e] = sum_c presence[n,c]*(x[n]@Wc[c])[e] + sum_c presence[n,c]*bc[c,e]
// launch_bounds(256, 4): cap VGPRs at 256 -> no vgpr-msb mode, >=4 waves/SIMD
__global__ __launch_bounds__(256, 4)
void guarded_gemm_kernel(const float* __restrict__ x,
                         const __bf16* __restrict__ WcT,   // [C][E][D] bf16
                         const float* __restrict__ bc,
                         const float* __restrict__ presence,
                         float* __restrict__ out)
{
    __shared__ __bf16 A_s[BM * ASTR];        // 12 KB  unscaled bf16 x tile (row-major, K contig)
    __shared__ float  p_s[BM * C_CASES];     //  4 KB  presence for this row block

    const int tid  = threadIdx.x;
    const int row0 = blockIdx.x * BM;
    const int col0 = blockIdx.y * BN;

    for (int i = tid; i < BM * C_CASES; i += 256)
        p_s[i] = presence[(size_t)row0 * C_CASES + i];

    const int lane  = tid & 31;
    const int wave  = tid >> 5;
    const int wm    = wave >> 2;     // 0..1  (64-row wave tile)
    const int wn    = wave & 3;      // 0..3  (32-col wave tile)
    const int khalf = lane >> 4;     // lanes 16-31 take the second K half
    const int lrow  = lane & 15;

    __syncthreads();

    v8f acc[4][2];
#pragma unroll
    for (int s = 0; s < 4; ++s)
#pragma unroll
        for (int t = 0; t < 2; ++t) acc[s][t] = {};

    const int colg0 = col0 + wn * 32 + lrow;         // fragment-0 column for this lane
    const int colg1 = colg0 + 16;                    // fragment-1 column
    // per-case stride through WcT is E*D elements
    const __bf16* bptr0 = WcT + (size_t)colg0 * D_DIM + khalf * 16;
    const __bf16* bptr1 = WcT + (size_t)colg1 * D_DIM + khalf * 16;

    for (int k0 = 0; k0 < D_DIM; k0 += BK) {
        __syncthreads();                             // prior A-frag reads done before restage
        // ---- stage unscaled bf16 x tile (BM x BK) ----
#pragma unroll
        for (int i = 0; i < 4; ++i) {
            const int slot = tid + i * 256;          // 1024 float4 slots
            const int r = slot >> 3, c4 = slot & 7;
            const float4 v = *(const float4*)&x[(size_t)(row0 + r) * D_DIM + k0 + c4 * 4];
            v4bf bb;
            bb[0] = (__bf16)v.x; bb[1] = (__bf16)v.y;
            bb[2] = (__bf16)v.z; bb[3] = (__bf16)v.w;
            *(v4bf*)&A_s[r * ASTR + c4 * 4] = bb;
        }
        __syncthreads();

        // ---- load raw A fragments once per k-step (contiguous k = 16*khalf + s, matching B) ----
        v16bf a_raw[4];
#pragma unroll
        for (int s = 0; s < 4; ++s) {
            const int r = wm * 64 + s * 16 + lrow;
            v8bf lo = *(const v8bf*)&A_s[r * ASTR + khalf * 16];
            v8bf hi = *(const v8bf*)&A_s[r * ASTR + khalf * 16 + 8];
            a_raw[s] = __builtin_shufflevector(lo, hi,
                        0,1,2,3,4,5,6,7,8,9,10,11,12,13,14,15);
        }

        // ---- software-pipelined case loop (unroll 1: <=2 B frag pairs live) ----
        v16bf b0 = *(const v16bf*)(bptr0 + k0);
        v16bf b1 = *(const v16bf*)(bptr1 + k0);
#pragma unroll 1
        for (int c = 0; c < C_CASES; ++c) {
            const int cn = (c + 1) & (C_CASES - 1);      // branchless wrap prefetch
            const v16bf nb0 = *(const v16bf*)(bptr0 + (size_t)cn * E_DIM * D_DIM + k0);
            const v16bf nb1 = *(const v16bf*)(bptr1 + (size_t)cn * E_DIM * D_DIM + k0);

            // presence-scale all four A fragments up front: 32 independent
            // v_pk_mul_bf16 that can fill the WMMA hazard windows (co-exec)
            v16bf as[4];
#pragma unroll
            for (int s = 0; s < 4; ++s) {
                const __bf16 pb = (__bf16)p_s[(wm * 64 + s * 16 + lrow) * C_CASES + c];
                as[s] = a_raw[s] * pb;
            }
#pragma unroll
            for (int s = 0; s < 4; ++s) {
                acc[s][0] = __builtin_amdgcn_wmma_f32_16x16x32_bf16(
                    false, as[s], false, b0, (short)0, acc[s][0], false, false);
                acc[s][1] = __builtin_amdgcn_wmma_f32_16x16x32_bf16(
                    false, as[s], false, b1, (short)0, acc[s][1], false, false);
            }
            b0 = nb0;
            b1 = nb1;
        }
    }

    // ---- epilogue: + presence @ bc, store f32 ----
    // C/D layout: VGPR v -> M = v + 8*khalf ; N = lane%16
    float bcv[2][C_CASES];
#pragma unroll
    for (int t = 0; t < 2; ++t) {
        const int colg = col0 + wn * 32 + t * 16 + lrow;
#pragma unroll
        for (int c = 0; c < C_CASES; ++c) bcv[t][c] = bc[c * E_DIM + colg];
    }
#pragma unroll
    for (int s = 0; s < 4; ++s) {
#pragma unroll
        for (int v = 0; v < 8; ++v) {
            const int rl = wm * 64 + s * 16 + khalf * 8 + v;
            const float4 pa = *(const float4*)&p_s[rl * C_CASES];
            const float4 pz = *(const float4*)&p_s[rl * C_CASES + 4];
#pragma unroll
            for (int t = 0; t < 2; ++t) {
                const float bias =
                    pa.x * bcv[t][0] + pa.y * bcv[t][1] + pa.z * bcv[t][2] + pa.w * bcv[t][3] +
                    pz.x * bcv[t][4] + pz.y * bcv[t][5] + pz.z * bcv[t][6] + pz.w * bcv[t][7];
                const int colg = col0 + wn * 32 + t * 16 + lrow;
                out[(size_t)(row0 + rl) * E_DIM + colg] = acc[s][t][v] + bias;
            }
        }
    }
}

extern "C" void kernel_launch(void* const* d_in, const int* in_sizes, int n_in,
                              void* d_out, int out_size, void* d_ws, size_t ws_size,
                              hipStream_t stream)
{
    const float* x  = (const float*)d_in[0];
    const float* Wp = (const float*)d_in[1];
    const float* bp = (const float*)d_in[2];
    const float* Wc = (const float*)d_in[3];
    const float* bc = (const float*)d_in[4];
    float* out      = (float*)d_out;

    const int N = in_sizes[0] / D_DIM;

    float*  presence = (float*)d_ws;                                     // N*C*4 = 2 MB
    __bf16* WcT      = (__bf16*)((char*)d_ws + (size_t)N * C_CASES * 4); // C*E*D*2 = 4 MB

    presence_kernel<<<N / 64, 256, 0, stream>>>(x, Wp, bp, presence);

    dim3 tgrid(E_DIM / 32, D_DIM / 32, C_CASES);
    wct_kernel<<<tgrid, 256, 0, stream>>>(Wc, WcT);

    dim3 grid(N / BM, E_DIM / BN);
    guarded_gemm_kernel<<<grid, 256, 0, stream>>>(x, WcT, bc, presence, out);
}
